// HGTModel_70514773066274
// MI455X (gfx1250) — matrix operand order repaired and checked by reading, not compiled, not run on hardware
//
#include <hip/hip_runtime.h>
#include <math.h>

// ---------------------------------------------------------------------------
// HGT forward on MI455X (gfx1250, wave32).
// GEMMs -> v_wmma_f32_16x16x32_bf16 (bf16 in, f32 accumulate), with the
//   weight panel pre-transposed to bf16 and staged into LDS via the gfx1250
//   async global->LDS path (ASYNCcnt) when available.
// Edge softmax/aggregation -> 3 gather/atomic passes per relation.
// Gate+residual+LayerNorm -> fused wave-per-row kernel.
// ---------------------------------------------------------------------------

typedef __attribute__((ext_vector_type(16))) __bf16 bf16x16;
typedef __attribute__((ext_vector_type(8)))  __bf16 bf16x8;
typedef __attribute__((ext_vector_type(8)))  float  floatx8;
typedef int v4i_ __attribute__((vector_size(16)));   // matches async-LDS builtin

#if defined(__gfx1250__) && \
    __has_builtin(__builtin_amdgcn_global_load_async_to_lds_b128) && \
    __has_builtin(__builtin_amdgcn_s_wait_asynccnt)
#define USE_ASYNC_LDS 1
#else
#define USE_ASYNC_LDS 0
#endif

__device__ __forceinline__ float gelu_tanh(float x) {
    const float c0 = 0.7978845608028654f;           // sqrt(2/pi)
    float x3 = x * x * x;
    return 0.5f * x * (1.0f + tanhf(c0 * (x + 0.044715f * x3)));
}

// order-preserving float<->uint encoding for atomic segment-max
__device__ __forceinline__ unsigned fenc(float f) {
    unsigned u = __float_as_uint(f);
    return u ^ ((u & 0x80000000u) ? 0xFFFFFFFFu : 0x80000000u);
}
__device__ __forceinline__ float fdec(unsigned u) {
    unsigned v = u ^ ((u & 0x80000000u) ? 0x80000000u : 0xFFFFFFFFu);
    return __uint_as_float(v);
}

// ---------------------------------------------------------------------------
// Weight pre-pass: W f32 [K,128] -> Wt bf16, transposed, row stride K+8
// (padding keeps the per-column LDS stride off a bank multiple).
// ---------------------------------------------------------------------------
__global__ void convert_w_k(const float* __restrict__ W, __bf16* __restrict__ Wt,
                            int K)
{
    const int c = blockIdx.x;                 // 0..127 output column
    const int SW = K + 8;
    for (int k = threadIdx.x; k < K; k += blockDim.x)
        Wt[(size_t)c * SW + k] = (__bf16)W[(size_t)k * 128 + c];
}

// expand [H=4,D=32,D=32] per-head mats into transposed bf16 block-diag weight
__global__ void expand_blockdiag_k(const float* __restrict__ a,
                                   __bf16* __restrict__ Wt)   // stride 136
{
    const int c = blockIdx.x;    // 0..127 column
    const int k = threadIdx.x;   // 0..127 K index
    const int hk = k >> 5, hc = c >> 5;
    float v = (hk == hc) ? a[hk * 1024 + (k & 31) * 32 + (c & 31)] : 0.0f;
    Wt[c * 136 + k] = (__bf16)v;
}

// ---------------------------------------------------------------------------
// Y[n,128] = act(X[n,K]) @ W[K,128] + bias   (bf16 WMMA, f32 accumulate)
// 256 threads = 8 waves; wave = 16 rows x 128 cols (8 accumulators).
// Wt (bf16, transposed, stride K+8) is staged once into LDS per block.
// ---------------------------------------------------------------------------
template <bool GELU_IN>
__global__ __launch_bounds__(256)
void gemm128_wmma(const float* __restrict__ X, const __bf16* __restrict__ Wt,
                  const float* __restrict__ bias, float* __restrict__ Y,
                  int nrows, int K)
{
    extern __shared__ __bf16 sW[];
    const int SW = K + 8;

    // ---- stage the whole weight panel (128*SW bf16) into LDS ----
    const int totalBytes = 128 * SW * 2;      // multiple of 16
    for (int ofs = threadIdx.x * 16; ofs < totalBytes; ofs += 256 * 16) {
#if USE_ASYNC_LDS
        __builtin_amdgcn_global_load_async_to_lds_b128(
            (__attribute__((address_space(1))) v4i_*)((char*)Wt + ofs),
            (__attribute__((address_space(3))) v4i_*)((char*)sW + ofs),
            0, 0);
#else
        *(float4*)((char*)sW + ofs) = *(const float4*)((const char*)Wt + ofs);
#endif
    }
#if USE_ASYNC_LDS
    __builtin_amdgcn_s_wait_asynccnt(0);
#endif
    __syncthreads();

    const int  lane = threadIdx.x & 31;
    const int  wave = threadIdx.x >> 5;
    const int  l16  = lane & 15;
    const bool hi   = lane >= 16;
    const int  row0 = blockIdx.x * 128 + wave * 16;
    const int  rowA = row0 + l16;
    const bool rowOK = rowA < nrows;
    const float* xrow = X + (size_t)rowA * K;

    floatx8 acc[8];
    #pragma unroll
    for (int i = 0; i < 8; ++i)
        #pragma unroll
        for (int j = 0; j < 8; ++j) acc[i][j] = 0.0f;

    for (int kb = 0; kb < K; kb += 32) {
        // A fragment (ISA 16-bit A 16x32 layout):
        // lanes 0-15: K = kb+{0..7,16..23}; lanes 16-31: K = kb+{8..15,24..31}
        const int k0 = kb + (hi ? 8 : 0);
        float4 x0 = {0.f,0.f,0.f,0.f}, x1 = x0, x2 = x0, x3 = x0;
        if (rowOK) {
            x0 = *(const float4*)(xrow + k0);
            x1 = *(const float4*)(xrow + k0 + 4);
            x2 = *(const float4*)(xrow + k0 + 16);
            x3 = *(const float4*)(xrow + k0 + 20);
        }
        float xv[16] = { x0.x, x0.y, x0.z, x0.w, x1.x, x1.y, x1.z, x1.w,
                         x2.x, x2.y, x2.z, x2.w, x3.x, x3.y, x3.z, x3.w };
        bf16x16 a;
        #pragma unroll
        for (int i = 0; i < 16; ++i) {
            float v = xv[i];
            if (GELU_IN) v = gelu_tanh(v);
            a[i] = (__bf16)v;
        }

        // B fragments from LDS: lane = column; lanes 0-15 carry K kb+{0..7,16..23},
        // lanes 16-31 carry K kb+{8..15,24..31} (2x ds_load_b128)
        const __bf16* wl0 = sW + kb + (hi ? 8 : 0);
        #pragma unroll
        for (int ct = 0; ct < 8; ++ct) {
            const int col = ct * 16 + l16;
            const __bf16* wl = wl0 + (size_t)col * SW;
            bf16x8 blo = *(const bf16x8*)(wl);
            bf16x8 bhi = *(const bf16x8*)(wl + 16);
            bf16x16 b;
            #pragma unroll
            for (int i = 0; i < 8; ++i) { b[i] = blo[i]; b[8 + i] = bhi[i]; }
            acc[ct] = __builtin_amdgcn_wmma_f32_16x16x32_bf16(
                false, a, false, b, (short)0, acc[ct], false, false);
        }
    }

    // D layout: lanes 0-15 -> rows row0+0..7, lanes 16-31 -> rows row0+8..15
    const int rbase = row0 + (hi ? 8 : 0);
    #pragma unroll
    for (int ct = 0; ct < 8; ++ct) {
        const int col = ct * 16 + l16;
        const float bv = bias ? bias[col] : 0.0f;
        #pragma unroll
        for (int i = 0; i < 8; ++i) {
            const int r = rbase + i;
            if (r < nrows) Y[(size_t)r * 128 + col] = acc[ct][i] + bv;
        }
    }
}

// pass A: logits[e,h] = <q[dst], k_rel[src]>_h * p[h] / sqrt(D); segment-max
__global__ __launch_bounds__(256)
void edge_logits_k(const int* __restrict__ ei, int E,
                   const float* __restrict__ Q, const float* __restrict__ Krel,
                   const float* __restrict__ p,
                   float* __restrict__ logits, unsigned* __restrict__ mx)
{
    const int tid = blockIdx.x * blockDim.x + threadIdx.x;
    if (tid >= E * 4) return;
    const int e = tid >> 2, h = tid & 3;
    const int si = ei[e], di = ei[E + e];
    const float* qv = Q    + (size_t)di * 128 + h * 32;
    const float* kv = Krel + (size_t)si * 128 + h * 32;
    float acc = 0.0f;
    #pragma unroll
    for (int d = 0; d < 32; ++d) acc += qv[d] * kv[d];
    acc *= p[h] * 0.17677669529663687f;   // 1/sqrt(32)
    logits[tid] = acc;
    atomicMax(&mx[di * 4 + h], fenc(acc));
}

// pass B: ex = exp(logit - max); segment-sum of ex
__global__ __launch_bounds__(256)
void edge_exp_k(const int* __restrict__ ei, int E, float* __restrict__ logits,
                const unsigned* __restrict__ mx, float* __restrict__ den)
{
    const int tid = blockIdx.x * blockDim.x + threadIdx.x;
    if (tid >= E * 4) return;
    const int e = tid >> 2, h = tid & 3;
    const int di = ei[E + e];
    const float ex = expf(logits[tid] - fdec(mx[di * 4 + h]));
    logits[tid] = ex;
    atomicAdd(&den[di * 4 + h], ex);
}

// pass C: agg[dst] += alpha * v_rel[src]
__global__ __launch_bounds__(256)
void edge_msg_k(const int* __restrict__ ei, int E,
                const float* __restrict__ logits, const float* __restrict__ den,
                const float* __restrict__ Vrel, float* __restrict__ agg)
{
    const int tid = blockIdx.x * blockDim.x + threadIdx.x;
    if (tid >= E * 4) return;
    const int e = tid >> 2, h = tid & 3;
    const int si = ei[e], di = ei[E + e];
    const float alpha = logits[tid] / (den[di * 4 + h] + 1e-16f);
    const float* vv = Vrel + (size_t)si * 128 + h * 32;
    float*       av = agg  + (size_t)di * 128 + h * 32;
    #pragma unroll
    for (int d = 0; d < 32; ++d) atomicAdd(&av[d], vv[d] * alpha);
}

// fused: out = LayerNorm(gate*o + (1-gate)*x + x) * g + be ; one wave per row
__global__ __launch_bounds__(256)
void mix_ln_k(const float* __restrict__ O, const float* __restrict__ Xin,
              const float* __restrict__ skip, const float* __restrict__ g,
              const float* __restrict__ be, float* __restrict__ out, int nrows)
{
    const int lane = threadIdx.x & 31;
    const int wave = threadIdx.x >> 5;
    const int row  = blockIdx.x * 8 + wave;
    if (row >= nrows) return;
    const float gate = 1.0f / (1.0f + expf(-skip[0]));

    float r[4];
    float s = 0.0f;
    #pragma unroll
    for (int j = 0; j < 4; ++j) {
        const int c = lane * 4 + j;
        const float o = O  [(size_t)row * 128 + c];
        const float x = Xin[(size_t)row * 128 + c];
        r[j] = gate * o + (1.0f - gate) * x + x;   // gated output + residual
        s += r[j];
    }
    #pragma unroll
    for (int m = 16; m > 0; m >>= 1) s += __shfl_xor(s, m, 32);
    const float mean = s * (1.0f / 128.0f);
    float v = 0.0f;
    #pragma unroll
    for (int j = 0; j < 4; ++j) { const float d = r[j] - mean; v += d * d; }
    #pragma unroll
    for (int m = 16; m > 0; m >>= 1) v += __shfl_xor(v, m, 32);
    const float rstd = rsqrtf(v * (1.0f / 128.0f) + 1e-5f);
    #pragma unroll
    for (int j = 0; j < 4; ++j) {
        const int c = lane * 4 + j;
        out[(size_t)row * 128 + c] = (r[j] - mean) * rstd * g[c] + be[c];
    }
}

// ---------------------------------------------------------------------------
// Host orchestration
// ---------------------------------------------------------------------------
extern "C" void kernel_launch(void* const* d_in, const int* in_sizes, int n_in,
                              void* d_out, int out_size, void* d_ws, size_t ws_size,
                              hipStream_t stream)
{
    const int NT[3]     = {100000, 50000, 20000};   // drug, se, bio
    const int OFFR[3]   = {0, 100000, 150000};
    const int INDIM[3]  = {256, 128, 64};
    const int TREE_T[3] = {1, 2, 0};                // type -> sorted(bio,drug,se) index
    const int NTOT      = 170000;

    auto F = [&](int i) -> const float* { return (const float*)d_in[i]; };

    // ---- workspace carve ----
    char* ws = (char*)d_ws;
    size_t cur = 0;
    auto carve = [&](size_t bytes) -> char* {
        char* p = ws + cur;
        cur = (cur + bytes + 255) & ~(size_t)255;
        return p;
    };
    float* Xb  = (float*)carve((size_t)NTOT * 128 * 4);
    float* Kb  = (float*)carve((size_t)NTOT * 128 * 4);
    float* Qb  = (float*)carve((size_t)NTOT * 128 * 4);
    float* Vb  = (float*)carve((size_t)NTOT * 128 * 4);
    float* AGG = (float*)carve((size_t)NTOT * 128 * 4);
    float* T0  = (float*)carve((size_t)100000 * 128 * 4);  // k_rel / gelu-proj out
    float* T1  = (float*)carve((size_t)100000 * 128 * 4);  // v_rel
    const int eiIdx[4] = {3, 4, 5, 6};
    int EMAX = 1;
    for (int i = 0; i < 4; ++i) { int e = in_sizes[eiIdx[i]] / 2; if (e > EMAX) EMAX = e; }
    float*    LG  = (float*)   carve((size_t)EMAX * 4 * 4);
    unsigned* MX  = (unsigned*)carve((size_t)100000 * 4 * 4);
    float*    DEN = (float*)   carve((size_t)100000 * 4 * 4);
    __bf16*   WTt = (__bf16*)  carve((size_t)128 * (256 + 8) * 2);  // generic weight
    __bf16*   WAt = (__bf16*)  carve((size_t)128 * 136 * 2);        // rel 'a' expanded
    __bf16*   WMt = (__bf16*)  carve((size_t)128 * 136 * 2);        // rel 'm' expanded

    auto gemmT = [&](const float* X, const __bf16* Wt, const float* b, float* Y,
                     int n, int K, bool gelu) {
        dim3 grid((n + 127) / 128), block(256);
        const size_t sh = (size_t)128 * (K + 8) * 2;
        if (gelu) gemm128_wmma<true ><<<grid, block, sh, stream>>>(X, Wt, b, Y, n, K);
        else      gemm128_wmma<false><<<grid, block, sh, stream>>>(X, Wt, b, Y, n, K);
    };
    auto gemm = [&](const float* X, const float* Wf, const float* b, float* Y,
                    int n, int K, bool gelu) {
        convert_w_k<<<dim3(128), dim3(256), 0, stream>>>(Wf, WTt, K);
        gemmT(X, WTt, b, Y, n, K, gelu);
    };

    // ---- param index map (jax tree_leaves order: layers then proj) ----
    // node arrays per type: Wa,Wk,Wq,Wv,ba,be,bk,bq,bv,g,skip (11)
    // rel arrays (b2d,d2b,d2s,s2d order): a,m,p (3 each)
    // layer block = 3*11 + 4*3 = 45 ; proj = W/b per type (bio,drug,se)

    // ---- input projections ----
    for (int t = 0; t < 3; ++t) {
        const int pW = 7 + 90 + TREE_T[t] * 2;
        gemm(F(t), F(pW), F(pW + 1), Xb + (size_t)OFFR[t] * 128, NT[t], INDIM[t], false);
    }

    struct Rel { int s, d, ei, rt; };
    const Rel rels[4] = { {0,1,3,2},   // d2s
                          {1,0,4,3},   // s2d
                          {0,2,5,1},   // d2b
                          {2,0,6,0} }; // b2d

    for (int l = 0; l < 2; ++l) {
        const int lb = 7 + l * 45;

        // K/Q/V projections for all types
        for (int t = 0; t < 3; ++t) {
            const int nb = lb + TREE_T[t] * 11;
            const float* x = Xb + (size_t)OFFR[t] * 128;
            gemm(x, F(nb + 1), F(nb + 6), Kb + (size_t)OFFR[t] * 128, NT[t], 128, false);
            gemm(x, F(nb + 2), F(nb + 7), Qb + (size_t)OFFR[t] * 128, NT[t], 128, false);
            gemm(x, F(nb + 3), F(nb + 8), Vb + (size_t)OFFR[t] * 128, NT[t], 128, false);
        }

        (void)hipMemsetAsync(AGG, 0, (size_t)NTOT * 128 * 4, stream);

        for (int r = 0; r < 4; ++r) {
            const Rel& R = rels[r];
            const int rb = lb + 33 + R.rt * 3;
            const int E  = in_sizes[R.ei] / 2;
            const int* ei = (const int*)d_in[R.ei];

            expand_blockdiag_k<<<128, 128, 0, stream>>>(F(rb + 0), WAt);
            expand_blockdiag_k<<<128, 128, 0, stream>>>(F(rb + 1), WMt);
            gemmT(Kb + (size_t)OFFR[R.s] * 128, WAt, nullptr, T0, NT[R.s], 128, false);
            gemmT(Vb + (size_t)OFFR[R.s] * 128, WMt, nullptr, T1, NT[R.s], 128, false);

            (void)hipMemsetAsync(MX,  0, (size_t)NT[R.d] * 4 * 4, stream);
            (void)hipMemsetAsync(DEN, 0, (size_t)NT[R.d] * 4 * 4, stream);

            const int nthr = E * 4;
            dim3 eg((nthr + 255) / 256), eb(256);
            edge_logits_k<<<eg, eb, 0, stream>>>(ei, E, Qb + (size_t)OFFR[R.d] * 128,
                                                 T0, F(rb + 2), LG, MX);
            edge_exp_k<<<eg, eb, 0, stream>>>(ei, E, LG, MX, DEN);
            edge_msg_k<<<eg, eb, 0, stream>>>(ei, E, LG, DEN, T1,
                                              AGG + (size_t)OFFR[R.d] * 128);
        }

        // output projection (gelu on input) + gate + residual + LayerNorm
        for (int t = 0; t < 3; ++t) {
            const int nb = lb + TREE_T[t] * 11;
            gemm(AGG + (size_t)OFFR[t] * 128, F(nb + 0), F(nb + 4), T0, NT[t], 128, true);
            float* outp = (l == 1) ? ((float*)d_out + (size_t)OFFR[t] * 128)
                                   : (Xb + (size_t)OFFR[t] * 128);
            dim3 mg((NT[t] + 7) / 8), mb(256);
            mix_ln_k<<<mg, mb, 0, stream>>>(T0, Xb + (size_t)OFFR[t] * 128,
                                            F(nb + 10), F(nb + 9), F(nb + 5),
                                            outp, NT[t]);
        }
    }
    (void)n_in; (void)out_size; (void)ws_size;
}
